// Gate_76003741270245
// MI455X (gfx1250) — matrix operand order repaired and checked by reading, not compiled, not run on hardware
//
#include <hip/hip_runtime.h>
#include <hip/hip_bf16.h>

typedef __attribute__((ext_vector_type(16))) __bf16 v16bf;
typedef __attribute__((ext_vector_type(8)))  float  v8f;
typedef __attribute__((ext_vector_type(4)))  float  v4f;

#define D_MODEL          4096
#define NUM_EXPERT       64
#define N_TOKENS         16384
#define TOKENS_PER_WAVE  16
#define WAVES_PER_BLOCK  8
#define TOKENS_PER_BLOCK (TOKENS_PER_WAVE * WAVES_PER_BLOCK)   // 128

// ---------------------------------------------------------------------------
// Prep: split fp32 W into hi/lo bf16 pair (x ~= hi + lo), stored [expert][K]
// so B fragments are contiguous 32B per lane. W is 1 MB total -> L2 resident.
// ---------------------------------------------------------------------------
__global__ void split_w_kernel(const float* __restrict__ W,
                               __bf16* __restrict__ Whi,
                               __bf16* __restrict__ Wlo, int n) {
    int i = blockIdx.x * blockDim.x + threadIdx.x;
    if (i < n) {
        float x = W[i];
        __bf16 h = (__bf16)x;
        Whi[i] = h;
        Wlo[i] = (__bf16)(x - (float)h);
    }
}

// ---------------------------------------------------------------------------
// Main: per wave, 16 tokens x 64 experts via v_wmma_f32_16x16x32_bf16 with
// hi/lo split (3 WMMAs per K=32 chunk per 16-expert group). Epilogue does
// bias + softmax-top2 through LDS.
// ---------------------------------------------------------------------------
__global__ __launch_bounds__(256) void moe_gate_kernel(
    const float*  __restrict__ inp,
    const __bf16* __restrict__ Whi,
    const __bf16* __restrict__ Wlo,
    const float*  __restrict__ bias,
    int*          __restrict__ idx_out,
    float*        __restrict__ score_out) {

    __shared__ float lds_logits[TOKENS_PER_BLOCK][NUM_EXPERT];  // 32 KB

    const int lane = threadIdx.x & 31;
    const int wave = threadIdx.x >> 5;
    const int m    = lane & 15;       // A row / B column within tile
    const int h    = lane >> 4;       // half-wave id
    const int tok0 = blockIdx.x * TOKENS_PER_BLOCK + wave * TOKENS_PER_WAVE;

    const float* __restrict__ arow = inp + (size_t)(tok0 + m) * D_MODEL;

    v8f acc[4];
#pragma unroll
    for (int g = 0; g < 4; ++g) acc[g] = (v8f)0.0f;

    for (int k = 0; k < D_MODEL; k += 32) {
        // Prefetch the A row ~3 chunks ahead (speculative; OOB is dropped).
        __builtin_prefetch(arow + k + 96, 0, 0);

        // ---- A fragment (16x32 bf16, contiguous K order matching B):
        //   j in [0,16): K = k + 16*h + j
        v4f av[4];
        av[0] = *(const v4f*)(arow + k + 16 * h);
        av[1] = *(const v4f*)(arow + k + 16 * h + 4);
        av[2] = *(const v4f*)(arow + k + 16 * h + 8);
        av[3] = *(const v4f*)(arow + k + 16 * h + 12);

        v16bf ahi, alo;
#pragma unroll
        for (int j = 0; j < 16; ++j) {
            float x = av[j >> 2][j & 3];
            __bf16 hb = (__bf16)x;
            ahi[j] = hb;
            alo[j] = (__bf16)(x - (float)hb);
        }

#pragma unroll
        for (int g = 0; g < 4; ++g) {
            // ---- B fragment (32x16 bf16): lane holds expert (16g+m),
            // K = k + 16*h + [0..16) -> contiguous 32 B in [expert][K] layout.
            const size_t boff = (size_t)(16 * g + m) * D_MODEL + k + 16 * h;
            v16bf bhi = *(const v16bf*)(Whi + boff);
            v16bf blo = *(const v16bf*)(Wlo + boff);

            acc[g] = __builtin_amdgcn_wmma_f32_16x16x32_bf16(
                false, ahi, false, bhi, (short)0, acc[g], false, false);
            acc[g] = __builtin_amdgcn_wmma_f32_16x16x32_bf16(
                false, ahi, false, blo, (short)0, acc[g], false, false);
            acc[g] = __builtin_amdgcn_wmma_f32_16x16x32_bf16(
                false, alo, false, bhi, (short)0, acc[g], false, false);
        }
    }

    // ---- Epilogue: C layout -> lane holds (token = v + 8*h, expert = 16g+m)
#pragma unroll
    for (int g = 0; g < 4; ++g) {
        float bg = bias[16 * g + m];
#pragma unroll
        for (int v = 0; v < 8; ++v) {
            int tl = wave * TOKENS_PER_WAVE + v + 8 * h;
            lds_logits[tl][16 * g + m] = acc[g][v] + bg;
        }
    }
    __syncthreads();

    // ---- Softmax + top-2 over 64 experts; one thread per token.
    if (threadIdx.x < TOKENS_PER_BLOCK) {
        const int t = threadIdx.x;
        const float* row = lds_logits[t];

        float mx = row[0];
#pragma unroll 8
        for (int e = 1; e < NUM_EXPERT; ++e) mx = fmaxf(mx, row[e]);

        float sum = 0.0f;
        float v1 = -3.402823466e+38f, v2 = -3.402823466e+38f;
        int   i1 = 0, i2 = 0;
#pragma unroll 4
        for (int e = 0; e < NUM_EXPERT; ++e) {
            float l = row[e];
            sum += __expf(l - mx);
            if (l > v1)      { v2 = v1; i2 = i1; v1 = l; i1 = e; }
            else if (l > v2) { v2 = l; i2 = e; }
        }
        float inv = 1.0f / sum;
        int gt = blockIdx.x * TOKENS_PER_BLOCK + t;
        idx_out[gt * 2 + 0]   = i1;
        idx_out[gt * 2 + 1]   = i2;
        score_out[gt * 2 + 0] = __expf(v1 - mx) * inv;
        score_out[gt * 2 + 1] = __expf(v2 - mx) * inv;
    }
}

// ---------------------------------------------------------------------------
extern "C" void kernel_launch(void* const* d_in, const int* in_sizes, int n_in,
                              void* d_out, int out_size, void* d_ws, size_t ws_size,
                              hipStream_t stream) {
    const float* inp  = (const float*)d_in[0];  // [N_TOKENS, D_MODEL]
    const float* W    = (const float*)d_in[1];  // [NUM_EXPERT, D_MODEL]
    const float* bias = (const float*)d_in[2];  // [NUM_EXPERT]

    __bf16* Whi = (__bf16*)d_ws;
    __bf16* Wlo = Whi + (size_t)NUM_EXPERT * D_MODEL;

    const int nW = NUM_EXPERT * D_MODEL;
    split_w_kernel<<<(nW + 255) / 256, 256, 0, stream>>>(W, Whi, Wlo, nW);

    // Tuple output: idx [N,2] int32 first, then score [N,2] f32.
    int*   idx_out   = (int*)d_out;
    float* score_out = (float*)d_out + (size_t)2 * N_TOKENS;

    moe_gate_kernel<<<N_TOKENS / TOKENS_PER_BLOCK, 256, 0, stream>>>(
        inp, Whi, Wlo, bias, idx_out, score_out);
}
